// hDS_RNN_37615323579036
// MI455X (gfx1250) — matrix operand order repaired and checked by reading, not compiled, not run on hardware
//
#include <hip/hip_runtime.h>
#include <cstdint>
#include <cstddef>

// ---------- types ----------
typedef __attribute__((ext_vector_type(16))) __bf16   v16bf;
typedef __attribute__((ext_vector_type(8)))  float    v8f;
typedef __attribute__((ext_vector_type(4)))  uint32_t u32x4;

// ---------- small helpers ----------
__device__ __forceinline__ uint16_t f2bf(float x) {
    __bf16 h = (__bf16)x;
    union { __bf16 h16; uint16_t u; } c; c.h16 = h; return c.u;
}

// A/B fragment load: 16 bf16 per lane, contiguous 32B -> two b128 loads.
__device__ __forceinline__ v16bf frag_lds(const uint16_t* pk, int kt, int lane) {
    const u32x4* p = (const u32x4*)(pk + kt * 512 + lane * 16);
    union { u32x4 q[2]; v16bf v; } u; u.q[0] = p[0]; u.q[1] = p[1]; return u.v;
}
__device__ __forceinline__ v16bf frag_g(const uint16_t* pk, size_t f, int lane) {
    const u32x4* p = (const u32x4*)(pk + (f * 32 + (size_t)lane) * 16);
    union { u32x4 q[2]; v16bf v; } u; u.q[0] = p[0]; u.q[1] = p[1]; return u.v;
}
__device__ __forceinline__ v8f wmma_bf16(v16bf a, v16bf b, v8f c) {
    return __builtin_amdgcn_wmma_f32_16x16x32_bf16(false, a, false, b, (short)0, c, false, false);
}
__device__ __forceinline__ float sigf(float x) { return 1.0f / (1.0f + __expf(-x)); }

// Store one scalar into packed A-fragment layout (16xK tile, bf16).
// A layout (ISA 7.12.2, 16-bit A 16x32): lane = M + 16*h; elem e: v=e>>1,p=e&1;
// K = v<4 ? 8h+2v+p : 16+8h+2(v-4)+p.
__device__ __forceinline__ void store_pk(uint16_t* pk, int M, int K, float val) {
    int kt = K >> 5, kk = K & 31;
    int h  = (kk >> 3) & 1;
    int vv = ((kk >> 4) << 2) + ((kk & 7) >> 1);
    int p  = kk & 1;
    pk[kt * 512 + (M + (h << 4)) * 16 + vv * 2 + p] = f2bf(val);
}

// ---------- prep kernels ----------
// Pack W (N rows x K cols, row-major f32, used as x @ W.T) into WMMA B fragments.
// B tile (32x16): lane L: N=L&15, K = 8*(L>>4) + e + (e&8).
__global__ __launch_bounds__(256) void pack_weight(const float* __restrict__ W,
                                                   uint16_t* __restrict__ out,
                                                   int N, int K) {
    int idx = blockIdx.x * 256 + threadIdx.x;
    int total = (K >> 5) * (N >> 4) * 32;
    if (idx >= total) return;
    int lane = idx & 31, f = idx >> 5;
    int NT = N >> 4;
    int kt = f / NT, nt = f % NT;
    int n = nt * 16 + (lane & 15);
    int kb = kt * 32 + 8 * (lane >> 4);
    uint16_t* o = out + (size_t)idx * 16;
    for (int e = 0; e < 16; e++) o[e] = f2bf(W[(size_t)n * K + kb + e + (e & 8)]);
}

// Pack x_t (per batch tile, per step) into A fragments: x[b][t][k], M=b-in-tile, K=k.
__global__ __launch_bounds__(256) void pack_x(const float* __restrict__ in0,
                                              uint16_t* __restrict__ xPk) {
    int idx = blockIdx.x * 256 + threadIdx.x;          // 64*128*4*32 = 1048576 total
    int lane = idx & 31, f = idx >> 5;
    int kt = f & 3, t = (f >> 2) & 127, tile = f >> 9;
    int hh = lane >> 4, M = lane & 15;
    int b = tile * 16 + M;
    uint16_t* o = xPk + (size_t)idx * 16;
    for (int e = 0; e < 16; e++) {
        int v = e >> 1, p = e & 1;
        int ko = (v < 4) ? (8 * hh + 2 * v + p) : (16 + 8 * hh + 2 * (v - 4) + p);
        int k = kt * 32 + ko;
        o[e] = f2bf(in0[((size_t)b * 128 + t) * 128 + k]);
    }
}

// ue_x[b,k,m] = sum_t inputs[b,t,k]*Ue_W[m,t] + Ue_b[m]  -- per-b WMMA GEMM.
__global__ __launch_bounds__(256) void uex_kernel(const float* __restrict__ in0,
                                                  const uint16_t* __restrict__ uePk,
                                                  const float* __restrict__ ueb,
                                                  __bf16* __restrict__ ue) {
    __shared__ float s_in[16384];                      // inputs[b]: 128t x 128k
    int b = blockIdx.x;
    for (int i = threadIdx.x; i < 16384; i += 256) s_in[i] = in0[(size_t)b * 16384 + i];
    __syncthreads();
    int w = threadIdx.x >> 5, lane = threadIdx.x & 31;
    int hh = lane >> 4, nlo = lane & 15;
    v16bf A[4];
    for (int kt = 0; kt < 4; kt++) {                   // A(M=k, K=t) = in[b][t][k]
        v16bf a;
        for (int e = 0; e < 16; e++) {
            int v = e >> 1, p = e & 1;
            int ko = (v < 4) ? (8 * hh + 2 * v + p) : (16 + 8 * hh + 2 * (v - 4) + p);
            int t = kt * 32 + ko;
            a[e] = (__bf16)s_in[t * 128 + (w * 16 + nlo)];
        }
        A[kt] = a;
    }
    for (int nt = 0; nt < 16; nt++) {
        v8f acc = {};
        for (int kt = 0; kt < 4; kt++)
            acc = wmma_bf16(A[kt], frag_g(uePk, (size_t)kt * 16 + nt, lane), acc);
        int n = nt * 16 + nlo;
        float bias = ueb[n];
        for (int r = 0; r < 8; r++) {
            int k = w * 16 + r + 8 * hh;
            ue[((size_t)b * 128 + k) * 256 + n] = (__bf16)(acc[r] + bias);
        }
    }
}

// ---------- main persistent kernel ----------
struct MainP {
    const float *in0;
    const float *Web, *Ue2b, *Veb, *Udb, *Wdb, *Vdb;
    const float *eBih, *eBhh, *mBih, *mBhh, *dBih, *dBhh;
    const float *VeW, *VdW, *regW, *regb;
    const uint16_t *wePk, *ue2Pk, *udPk, *wdPk;
    const uint16_t *eWihPk, *eWhhPk, *mWihPk, *mWhhPk, *dWihPk, *dWhhPk, *xPk;
    const __bf16 *ue;
    __bf16 *udm, *midp;
    float *out;
};

// One LSTM step: gates via WMMA (wave w owns gate cols [w*32, w*32+32) of each of
// i/f/g/o -> gates stay in C fragments), then pointwise update.
__device__ __forceinline__ void lstm_step(
    const uint16_t* xpk, int xkt, const uint16_t* hpk,
    const uint16_t* WihPk, const uint16_t* WhhPk,
    const float* bih, const float* bhh,
    float* cf, uint16_t* out_hpk, uint16_t* out_cpk,
    float* out_hf, __bf16* out_hg,
    int lane, int w, int hh, int nlo)
{
    v8f acc[8] = {};
    for (int kt = 0; kt < xkt; kt++) {
        v16bf a = frag_lds(xpk, kt, lane);
#pragma unroll
        for (int j = 0; j < 8; j++) {
            int nt = (j >> 1) * 16 + w * 2 + (j & 1);
            acc[j] = wmma_bf16(a, frag_g(WihPk, (size_t)kt * 64 + nt, lane), acc[j]);
        }
    }
    for (int kt = 0; kt < 8; kt++) {
        v16bf a = frag_lds(hpk, kt, lane);
#pragma unroll
        for (int j = 0; j < 8; j++) {
            int nt = (j >> 1) * 16 + w * 2 + (j & 1);
            acc[j] = wmma_bf16(a, frag_g(WhhPk, (size_t)kt * 64 + nt, lane), acc[j]);
        }
    }
    __syncthreads();                                    // all reads of hpk done
#pragma unroll
    for (int j2 = 0; j2 < 2; j2++) {
        int n = w * 32 + j2 * 16 + nlo;
        float bi = bih[n]       + bhh[n];
        float bf = bih[256 + n] + bhh[256 + n];
        float bg = bih[512 + n] + bhh[512 + n];
        float bo = bih[768 + n] + bhh[768 + n];
#pragma unroll
        for (int r = 0; r < 8; r++) {
            int M = r + 8 * hh;
            float gi = sigf(acc[0 + j2][r] + bi);
            float gf = sigf(acc[2 + j2][r] + bf);
            float gg = tanhf(acc[4 + j2][r] + bg);
            float go = sigf(acc[6 + j2][r] + bo);
            float c2 = gf * cf[M * 256 + n] + gi * gg;
            float h2 = go * tanhf(c2);
            cf[M * 256 + n] = c2;
            store_pk(out_hpk, M, n, h2);
            if (out_cpk) store_pk(out_cpk, M, n, c2);
            if (out_hf)  out_hf[M * 256 + n] = h2;
            if (out_hg)  out_hg[M * 256 + n] = (__bf16)h2;
        }
    }
    __syncthreads();
}

__global__ __launch_bounds__(256, 1) void dsrnn_main(MainP P) {
    __shared__ __align__(16) uint16_t s_ehpk[4096];     // enc h packed (8 kt)
    __shared__ __align__(16) uint16_t s_ecpk[4096];     // enc c packed
    __shared__ __align__(16) uint16_t s_mhpk[4096];     // mid h packed
    __shared__ __align__(16) uint16_t s_dhpk[4096];     // dec h packed
    __shared__ __align__(16) uint16_t s_dcpk[4096];     // dec c packed
    __shared__ __align__(16) uint16_t s_xinpk[2048];    // x_in packed (4 kt)
    __shared__ __align__(16) uint16_t s_dinpk[4096];    // dec_in packed (8 kt)
    __shared__ float s_cenc[4096], s_cmid[4096], s_cdec[4096];
    __shared__ float s_sum[4096];                       // 16x256 pre-tanh term
    __shared__ float s_hf[4096];                        // dec h fp32
    __shared__ float s_sc[2048];                        // 16x128 score / tvec
    __shared__ float s_ve[256], s_vd[256], s_reg[256];

    int tid = threadIdx.x, w = tid >> 5, lane = tid & 31;
    int hh = lane >> 4, nlo = lane & 15;
    int tile = blockIdx.x;

    for (int i = tid; i < 4096; i += 256) {
        s_ehpk[i] = 0; s_ecpk[i] = 0; s_mhpk[i] = 0; s_dhpk[i] = 0; s_dcpk[i] = 0;
        s_cenc[i] = 0.f; s_cmid[i] = 0.f; s_cdec[i] = 0.f;
    }
    s_ve[tid] = P.VeW[tid]; s_vd[tid] = P.VdW[tid]; s_reg[tid] = P.regW[tid];
    __syncthreads();
    float veb = P.Veb[0], vdb = P.Vdb[0], regb = P.regb[0];

    // ================= encoder + mid (fused), 128 steps =================
    for (int t = 0; t < 128; t++) {
        // s_sum = [h|c]@We.T + x_t@Ue2.T + (We_b+Ue2_b)
        v8f s0 = {}, s1 = {};
        for (int kt = 0; kt < 8; kt++) {
            v16bf a = frag_lds(s_ehpk, kt, lane);
            s0 = wmma_bf16(a, frag_g(P.wePk, (size_t)kt * 16 + w * 2 + 0, lane), s0);
            s1 = wmma_bf16(a, frag_g(P.wePk, (size_t)kt * 16 + w * 2 + 1, lane), s1);
        }
        for (int kt = 0; kt < 8; kt++) {
            v16bf a = frag_lds(s_ecpk, kt, lane);
            s0 = wmma_bf16(a, frag_g(P.wePk, (size_t)(kt + 8) * 16 + w * 2 + 0, lane), s0);
            s1 = wmma_bf16(a, frag_g(P.wePk, (size_t)(kt + 8) * 16 + w * 2 + 1, lane), s1);
        }
        const uint16_t* xf = P.xPk + ((size_t)(tile * 128 + t) * 4) * 512;
        for (int kt = 0; kt < 4; kt++) {
            v16bf a = frag_g(xf, kt, lane);
            s0 = wmma_bf16(a, frag_g(P.ue2Pk, (size_t)kt * 16 + w * 2 + 0, lane), s0);
            s1 = wmma_bf16(a, frag_g(P.ue2Pk, (size_t)kt * 16 + w * 2 + 1, lane), s1);
        }
        {
            int n0 = w * 32 + nlo, n1 = n0 + 16;
            float b0 = P.Web[n0] + P.Ue2b[n0], b1 = P.Web[n1] + P.Ue2b[n1];
            for (int r = 0; r < 8; r++) {
                int M = r + 8 * hh;
                s_sum[M * 256 + n0] = s0[r] + b0;
                s_sum[M * 256 + n1] = s1[r] + b1;
            }
        }
        __syncthreads();
        // attention score: sc[row,k] = Ve_b + sum_m Ve[m]*tanh(sum + ue_x)
        // Register-blocked over m: 8 independent tanh/FMA chains (one per k)
        // to hide trans-op latency; s_sum/Ve loaded once per m-block.
        {
            int row = tid >> 4, kb = (tid & 15) * 8;
            const __bf16* ubase = P.ue + ((size_t)(tile * 16 + row) * 128 + kb) * 256;
            float acc[8];
#pragma unroll
            for (int kk = 0; kk < 8; kk++) acc[kk] = veb;
            for (int m = 0; m < 256; m += 8) {
                float ss[8], vv[8];
#pragma unroll
                for (int j = 0; j < 8; j++) {
                    ss[j] = s_sum[row * 256 + m + j];
                    vv[j] = s_ve[m + j];
                }
#pragma unroll
                for (int kk = 0; kk < 8; kk++) {
                    union { u32x4 q; __bf16 e[8]; } u;
                    u.q = *(const u32x4*)(ubase + (size_t)kk * 256 + m);
#pragma unroll
                    for (int j = 0; j < 8; j++)
                        acc[kk] += vv[j] * tanhf(ss[j] + (float)u.e[j]);
                }
            }
#pragma unroll
            for (int kk = 0; kk < 8; kk++) s_sc[row * 128 + kb + kk] = acc[kk];
        }
        __syncthreads();
        // softmax over k (per row)
        if (tid < 16) {
            int row = tid;
            float mx = -1e30f;
            for (int k = 0; k < 128; k++) mx = fmaxf(mx, s_sc[row * 128 + k]);
            float sum = 0.f;
            for (int k = 0; k < 128; k++) {
                float e = __expf(s_sc[row * 128 + k] - mx);
                s_sc[row * 128 + k] = e; sum += e;
            }
            float inv = 1.f / sum;
            for (int k = 0; k < 128; k++) s_sc[row * 128 + k] *= inv;
        }
        __syncthreads();
        // x_in = x_t * a -> packed A
        {
            int row = tid >> 4, kb = (tid & 15) * 8;
            for (int kk = 0; kk < 8; kk++) {
                int k = kb + kk;
                float xv = P.in0[((size_t)(tile * 16 + row) * 128 + t) * 128 + k]
                           * s_sc[row * 128 + k];
                store_pk(s_xinpk, row, k, xv);
            }
        }
        __syncthreads();
        // encoder LSTM
        lstm_step(s_xinpk, 4, s_ehpk, P.eWihPk, P.eWhhPk, P.eBih, P.eBhh,
                  s_cenc, s_ehpk, s_ecpk, nullptr, nullptr, lane, w, hh, nlo);
        // mid LSTM (x = enc h2); also store mid_h plain bf16 for decoder einsum
        lstm_step(s_ehpk, 8, s_mhpk, P.mWihPk, P.mWhhPk, P.mBih, P.mBhh,
                  s_cmid, s_mhpk, nullptr, nullptr,
                  P.midp + ((size_t)(tile * 128 + t) * 16) * 256, lane, w, hh, nlo);
        // ud_mid[t] = mid_h @ Ud.T + Ud_b  (bf16 to workspace)
        {
            v8f u0 = {}, u1 = {};
            for (int kt = 0; kt < 8; kt++) {
                v16bf a = frag_lds(s_mhpk, kt, lane);
                u0 = wmma_bf16(a, frag_g(P.udPk, (size_t)kt * 16 + w * 2 + 0, lane), u0);
                u1 = wmma_bf16(a, frag_g(P.udPk, (size_t)kt * 16 + w * 2 + 1, lane), u1);
            }
            int n0 = w * 32 + nlo, n1 = n0 + 16;
            float b0 = P.Udb[n0], b1 = P.Udb[n1];
            __bf16* up = P.udm + ((size_t)(tile * 128 + t) * 16) * 256;
            for (int r = 0; r < 8; r++) {
                int M = r + 8 * hh;
                up[M * 256 + n0] = (__bf16)(u0[r] + b0);
                up[M * 256 + n1] = (__bf16)(u1[r] + b1);
            }
        }
    }
    __syncthreads();
    __threadfence();   // make udm/midp writes visible to all waves in this block

    // ================= decoder, 30 steps =================
    for (int st = 0; st < 30; st++) {
        // wd = [hd|cd]@Wd.T + Wd_b
        v8f s0 = {}, s1 = {};
        for (int kt = 0; kt < 8; kt++) {
            v16bf a = frag_lds(s_dhpk, kt, lane);
            s0 = wmma_bf16(a, frag_g(P.wdPk, (size_t)kt * 16 + w * 2 + 0, lane), s0);
            s1 = wmma_bf16(a, frag_g(P.wdPk, (size_t)kt * 16 + w * 2 + 1, lane), s1);
        }
        for (int kt = 0; kt < 8; kt++) {
            v16bf a = frag_lds(s_dcpk, kt, lane);
            s0 = wmma_bf16(a, frag_g(P.wdPk, (size_t)(kt + 8) * 16 + w * 2 + 0, lane), s0);
            s1 = wmma_bf16(a, frag_g(P.wdPk, (size_t)(kt + 8) * 16 + w * 2 + 1, lane), s1);
        }
        {
            int n0 = w * 32 + nlo, n1 = n0 + 16;
            float b0 = P.Wdb[n0], b1 = P.Wdb[n1];
            for (int r = 0; r < 8; r++) {
                int M = r + 8 * hh;
                s_sum[M * 256 + n0] = s0[r] + b0;
                s_sum[M * 256 + n1] = s1[r] + b1;
            }
        }
        __syncthreads();
        // tvec[row,t] = Vd_b + sum_m Vd[m]*tanh(wd + ud_mid)  (register-blocked)
        {
            int row = tid >> 4, tb = (tid & 15) * 8;
            const __bf16* ubase = P.udm + ((size_t)(tile * 128 + tb) * 16 + row) * 256;
            float acc[8];
#pragma unroll
            for (int tt = 0; tt < 8; tt++) acc[tt] = vdb;
            for (int m = 0; m < 256; m += 8) {
                float ss[8], vv[8];
#pragma unroll
                for (int j = 0; j < 8; j++) {
                    ss[j] = s_sum[row * 256 + m + j];
                    vv[j] = s_vd[m + j];
                }
#pragma unroll
                for (int tt = 0; tt < 8; tt++) {
                    union { u32x4 q; __bf16 e[8]; } u;
                    u.q = *(const u32x4*)(ubase + (size_t)tt * 16 * 256 + m);
#pragma unroll
                    for (int j = 0; j < 8; j++)
                        acc[tt] += vv[j] * tanhf(ss[j] + (float)u.e[j]);
                }
            }
#pragma unroll
            for (int tt = 0; tt < 8; tt++) s_sc[row * 128 + tb + tt] = acc[tt];
        }
        __syncthreads();
        // dec_in[row,h] = sum_t tvec[row,t]*mid[row,t,h]  (per-row B -> VALU)
        {
            int row = tid >> 4, nb = (tid & 15) * 16;
            float av[16];
            for (int j = 0; j < 16; j++) av[j] = 0.f;
            for (int tq = 0; tq < 128; tq++) {
                float tv = s_sc[row * 128 + tq];
                const __bf16* mp = P.midp + ((size_t)(tile * 128 + tq) * 16 + row) * 256 + nb;
                union { u32x4 q[2]; __bf16 e[16]; } u;
                u.q[0] = *(const u32x4*)mp;
                u.q[1] = *(const u32x4*)(mp + 8);
                for (int j = 0; j < 16; j++) av[j] += tv * (float)u.e[j];
            }
            for (int j = 0; j < 16; j++) store_pk(s_dinpk, row, nb + j, av[j]);
        }
        __syncthreads();
        // decoder LSTM (also produce fp32 h for regression dot)
        lstm_step(s_dinpk, 8, s_dhpk, P.dWihPk, P.dWhhPk, P.dBih, P.dBhh,
                  s_cdec, s_dhpk, s_dcpk, s_hf, nullptr, lane, w, hh, nlo);
        // out = h @ reg_W.T + reg_b, keep last 24 steps
        if (st >= 6 && tid < 16) {
            int row = tid;
            float acc = regb;
            for (int n = 0; n < 256; n++) acc += s_reg[n] * s_hf[row * 256 + n];
            P.out[(size_t)(tile * 16 + row) * 24 + (st - 6)] = acc;
        }
        __syncthreads();
    }
}

// ---------- host launcher ----------
extern "C" void kernel_launch(void* const* d_in, const int* in_sizes, int n_in,
                              void* d_out, int out_size, void* d_ws, size_t ws_size,
                              hipStream_t stream) {
    (void)in_sizes; (void)n_in; (void)out_size; (void)ws_size;
    const float* in0  = (const float*)d_in[0];
    const float* UeW  = (const float*)d_in[2];  const float* Ueb  = (const float*)d_in[3];
    const float* Ue2W = (const float*)d_in[4];  const float* Ue2b = (const float*)d_in[5];
    const float* WeW  = (const float*)d_in[6];  const float* Web  = (const float*)d_in[7];
    const float* VeW  = (const float*)d_in[8];  const float* Veb  = (const float*)d_in[9];
    const float* UdW  = (const float*)d_in[10]; const float* Udb  = (const float*)d_in[11];
    const float* WdW  = (const float*)d_in[12]; const float* Wdb  = (const float*)d_in[13];
    const float* VdW  = (const float*)d_in[14]; const float* Vdb  = (const float*)d_in[15];
    const float* eWih = (const float*)d_in[16]; const float* eWhh = (const float*)d_in[17];
    const float* eBih = (const float*)d_in[18]; const float* eBhh = (const float*)d_in[19];
    const float* mWih = (const float*)d_in[20]; const float* mWhh = (const float*)d_in[21];
    const float* mBih = (const float*)d_in[22]; const float* mBhh = (const float*)d_in[23];
    const float* dWih = (const float*)d_in[24]; const float* dWhh = (const float*)d_in[25];
    const float* dBih = (const float*)d_in[26]; const float* dBhh = (const float*)d_in[27];
    const float* regW = (const float*)d_in[28]; const float* regb = (const float*)d_in[29];

    uint8_t* ws = (uint8_t*)d_ws;
    size_t off = 0;
    auto take = [&](size_t n) -> uint8_t* {
        uint8_t* p = ws + off; off += (n + 255) & ~(size_t)255; return p;
    };
    auto wb = [](int N, int K) -> size_t { return (size_t)(K / 32) * (N / 16) * 1024; };

    uint16_t* uePk   = (uint16_t*)take(wb(256, 128));
    uint16_t* ue2Pk  = (uint16_t*)take(wb(256, 128));
    uint16_t* wePk   = (uint16_t*)take(wb(256, 512));
    uint16_t* udPk   = (uint16_t*)take(wb(256, 256));
    uint16_t* wdPk   = (uint16_t*)take(wb(256, 512));
    uint16_t* eWihPk = (uint16_t*)take(wb(1024, 128));
    uint16_t* eWhhPk = (uint16_t*)take(wb(1024, 256));
    uint16_t* mWihPk = (uint16_t*)take(wb(1024, 256));
    uint16_t* mWhhPk = (uint16_t*)take(wb(1024, 256));
    uint16_t* dWihPk = (uint16_t*)take(wb(1024, 256));
    uint16_t* dWhhPk = (uint16_t*)take(wb(1024, 256));
    uint16_t* xPk    = (uint16_t*)take((size_t)64 * 128 * 4 * 1024);
    __bf16*   ue     = (__bf16*)take((size_t)1024 * 128 * 256 * 2);
    __bf16*   udm    = (__bf16*)take((size_t)1024 * 128 * 256 * 2);
    __bf16*   midp   = (__bf16*)take((size_t)1024 * 128 * 256 * 2);

    auto packW = [&](const float* W, uint16_t* o, int N, int K) {
        int total = (K / 32) * (N / 16) * 32;
        pack_weight<<<(total + 255) / 256, 256, 0, stream>>>(W, o, N, K);
    };
    packW(UeW,  uePk,   256, 128);
    packW(Ue2W, ue2Pk,  256, 128);
    packW(WeW,  wePk,   256, 512);
    packW(UdW,  udPk,   256, 256);
    packW(WdW,  wdPk,   256, 512);
    packW(eWih, eWihPk, 1024, 128);
    packW(eWhh, eWhhPk, 1024, 256);
    packW(mWih, mWihPk, 1024, 256);
    packW(mWhh, mWhhPk, 1024, 256);
    packW(dWih, dWihPk, 1024, 256);
    packW(dWhh, dWhhPk, 1024, 256);

    pack_x<<<4096, 256, 0, stream>>>(in0, xPk);
    uex_kernel<<<1024, 256, 0, stream>>>(in0, uePk, Ueb, ue);

    MainP P;
    P.in0 = in0;
    P.Web = Web; P.Ue2b = Ue2b; P.Veb = Veb; P.Udb = Udb; P.Wdb = Wdb; P.Vdb = Vdb;
    P.eBih = eBih; P.eBhh = eBhh; P.mBih = mBih; P.mBhh = mBhh; P.dBih = dBih; P.dBhh = dBhh;
    P.VeW = VeW; P.VdW = VdW; P.regW = regW; P.regb = regb;
    P.wePk = wePk; P.ue2Pk = ue2Pk; P.udPk = udPk; P.wdPk = wdPk;
    P.eWihPk = eWihPk; P.eWhhPk = eWhhPk; P.mWihPk = mWihPk; P.mWhhPk = mWhhPk;
    P.dWihPk = dWihPk; P.dWhhPk = dWhhPk; P.xPk = xPk;
    P.ue = ue; P.udm = udm; P.midp = midp;
    P.out = (float*)d_out;

    dsrnn_main<<<64, 256, 0, stream>>>(P);
}
